// QHNet_backbone_MADFT_94489281041
// MI455X (gfx1250) — matrix operand order repaired and checked by reading, not compiled, not run on hardware
//
#include <hip/hip_runtime.h>
#include <cstdint>

// ---------------------------------------------------------------------------
// CDNA5 (gfx1250) QHNet backbone forward.
// bf16 WMMA (v_wmma_f32_16x16x32_bf16) for all GEMMs, fused message scatter,
// TDM (tensor_load_to_lds) weight staging into LDS for the node-update GEMMs.
// ---------------------------------------------------------------------------

typedef unsigned short u16;
typedef unsigned int   u32;
typedef __attribute__((ext_vector_type(16))) __bf16 v16bf;
typedef __attribute__((ext_vector_type(8)))  float  v8f;
typedef __attribute__((ext_vector_type(4)))  unsigned int v4u;
typedef __attribute__((ext_vector_type(8)))  int v8i;
typedef __attribute__((ext_vector_type(4)))  int v4i;

#define NNODES 16384
#define NEDGES 262144
#define CCH    128
#define KRBF   32
#define NLAYER 5
#define LOG2F_ 0.6931471805599453f
#define SQRT3F 1.7320508075688772f

#if defined(__has_builtin)
#if __has_builtin(__builtin_amdgcn_tensor_load_to_lds) && \
    __has_builtin(__builtin_amdgcn_s_wait_tensorcnt)
#define USE_TDM 1
#endif
#endif
#ifndef USE_TDM
#define USE_TDM 0
#endif

union FragBF { v16bf v; u32 u[8]; uint4 q[2]; };

__device__ __forceinline__ u16 bf16_rne(float f) {
  u32 u = __float_as_uint(f);
  u32 r = u + 0x7FFFu + ((u >> 16) & 1u);
  return (u16)(r >> 16);
}
__device__ __forceinline__ u32 pack2(float a, float b) {
  return (u32)bf16_rne(a) | ((u32)bf16_rne(b) << 16);
}
// softplus(x), numerically stable
__device__ __forceinline__ float sp(float x) {
  return fmaxf(x, 0.0f) + log1pf(__expf(-fabsf(x)));
}
// shifted softplus: softplus(x) - log(2)
__device__ __forceinline__ float ssp(float x) { return sp(x) - LOG2F_; }

__device__ __forceinline__ v8f wmma_bf16(const FragBF& a, const FragBF& b, v8f c) {
  return __builtin_amdgcn_wmma_f32_16x16x32_bf16(false, a.v, false, b.v,
                                                 (short)0, c, false, false);
}

// A-fragment (16x32, bf16) from row-major bf16 matrix: two b128 loads.
// lane layout: VGPR q holds K = (q>>2)*16 + hi*8 + (q&3)*2 (+1)
__device__ __forceinline__ FragBF load_a_bf16(const u16* __restrict__ M,
                                              size_t rowbase, int k0, int hi) {
  FragBF a;
  const u16* p = M + rowbase + k0 + (hi << 3);
  a.q[0] = *(const uint4*)(p);        // K = k0+hi*8   .. +7
  a.q[1] = *(const uint4*)(p + 16);   // K = k0+16+hi*8.. +7
  return a;
}
// A-fragment from fp32 source with scale; stride = element stride along K.
__device__ __forceinline__ FragBF load_a_f32(const float* __restrict__ row,
                                             int k0, int hi, float scale, int stride) {
  FragBF a;
#pragma unroll
  for (int q = 0; q < 8; ++q) {
    int k = k0 + ((q >> 2) << 4) + (hi << 3) + ((q & 3) << 1);
    float x0 = row[(size_t)k * stride] * scale;
    float x1 = row[(size_t)(k + 1) * stride] * scale;
    a.u[q] = pack2(x0, x1);
  }
  return a;
}
// B-fragment (32x16, bf16) from column-major weights W[col*ldk + k]:
// lane layout: VGPR q holds K = (q>>2)*16 + hi*8 + (q&3)*2 (+1)
__device__ __forceinline__ FragBF load_b_cm(const u16* __restrict__ Wcm,
                                            int col, int ldk, int k0, int hi) {
  FragBF b;
  const u16* p = Wcm + (size_t)col * ldk + k0 + (hi << 3);
  b.q[0] = *(const uint4*)(p);
  b.q[1] = *(const uint4*)(p + 16);
  return b;
}

// ---- Tensor Data Mover: 1-D bf16 copy global -> LDS (D# per ISA 8.3/8.4) ---
__device__ __forceinline__ void tdm_load_1d(const u16* gptr, u32 lds_off,
                                            u32 nelem) {
#if USE_TDM
  unsigned long long ga = (unsigned long long)(uintptr_t)gptr;
  v4u g0;
  g0[0] = 1u;                                       // count=1, user descriptor
  g0[1] = lds_off;                                  // lds_addr (bytes)
  g0[2] = (u32)(ga & 0xFFFFFFFFu);                  // global_addr[31:0]
  g0[3] = (u32)((ga >> 32) & 0x01FFFFFFu) | (2u << 30); // addr[56:32], type=2
  v8i g1 = {0, 0, 0, 0, 0, 0, 0, 0};
  g1[0] = (int)(1u << 16);                          // data_size = 2 bytes
  g1[1] = (int)((nelem & 0xFFFFu) << 16);           // tensor_dim0[15:0]
  g1[2] = (int)((nelem >> 16) & 0xFFFFu) | (1 << 16); // dim0[31:16], dim1=1
  g1[3] = (int)((nelem & 0xFFFFu) << 16);           // tile_dim0
  g1[4] = 1;                                        // tile_dim1 = 1
  g1[5] = (int)nelem;                               // tensor_dim0_stride
  v4i z4 = {0, 0, 0, 0};
  v8i z8 = {0, 0, 0, 0, 0, 0, 0, 0};
  // amdgpu-toolchain (clang-23) 6-arg form:
  // (uint32x4 g0, int32x8 g1, int32x4 g2, int32x4 g3, int32x8, i32 cpol)
  __builtin_amdgcn_tensor_load_to_lds(g0, g1, z4, z4, z8, 0);
#else
  (void)gptr; (void)lds_off; (void)nelem;
#endif
}

// ------------------------- prep kernels ------------------------------------

// fp32 row-major (Kdim x Ncols) -> bf16 column-major [n*Kdim + k]
__global__ void k_repack_cm(const float* __restrict__ W, u16* __restrict__ out,
                            int Kdim, int Ncols) {
  int id = blockIdx.x * blockDim.x + threadIdx.x;
  if (id >= Kdim * Ncols) return;
  int k = id / Ncols, n = id - k * Ncols;
  out[(size_t)n * Kdim + k] = bf16_rne(W[id]);
}

// per-edge geometry: rbf (bf16, E x 32) and y1 (fp32, E x 3)
__global__ void k_edge_geom(const float* __restrict__ pos,
                            const int* __restrict__ eidx,
                            const float* __restrict__ alpha_raw,
                            u16* __restrict__ rbf, float* __restrict__ y1) {
  int e = blockIdx.x * blockDim.x + threadIdx.x;
  if (e >= NEDGES) return;
  int si = eidx[e], di = eidx[NEDGES + e];
  float ex = pos[si * 3 + 0] - pos[di * 3 + 0];
  float ey = pos[si * 3 + 1] - pos[di * 3 + 1];
  float ez = pos[si * 3 + 2] - pos[di * 3 + 2];
  float r = sqrtf(ex * ex + ey * ey + ez * ez);
  float alpha = sp(alpha_raw[0]);
  float ar = alpha * r;
  float logomx = log1pf(-__expf(-ar));
  float ir = 1.0f / r;
  // y1 = sqrt(3) * u[:, [1,2,0]]
  y1[e * 3 + 0] = SQRT3F * ey * ir;
  y1[e * 3 + 1] = SQRT3F * ez * ir;
  y1[e * 3 + 2] = SQRT3F * ex * ir;
  float comb = 1.0f; // C(31, i)
#pragma unroll 1
  for (int i = 0; i < KRBF; ++i) {
    float logc = logf(comb);
    float vv = (float)(KRBF - 1 - i);
    float t = logc - (float)i * ar + ((vv == 0.0f) ? 0.0f : vv * logomx);
    rbf[(size_t)e * KRBF + i] = bf16_rne(__expf(t));
    comb = comb * (float)(KRBF - 1 - i) / (float)(i + 1);
  }
}

__global__ void k_count(const int* __restrict__ eidx, float* __restrict__ cnt) {
  int e = blockIdx.x * blockDim.x + threadIdx.x;
  if (e < NEDGES) atomicAdd(&cnt[eidx[NEDGES + e]], 1.0f);
}
__global__ void k_invcnt(const float* __restrict__ cnt, float* __restrict__ inv) {
  int n = blockIdx.x * blockDim.x + threadIdx.x;
  if (n < NNODES) inv[n] = 1.0f / fmaxf(cnt[n], 1.0f);
}
__global__ void k_init_s(const float* __restrict__ embed, const int* __restrict__ an,
                         float* __restrict__ s) {
  int id = blockIdx.x * blockDim.x + threadIdx.x;
  int n = id >> 7, c = id & 127;
  if (n < NNODES) s[id] = embed[an[n] * CCH + c];
}

// ------------------------- layer kernels -----------------------------------

// h = ssp(rbf @ W1 + b1) : one wave per 16-edge tile, 4 WMMAs (N=64)
__global__ __launch_bounds__(128) void k_edge_mlp1(
    const u16* __restrict__ rbf, const u16* __restrict__ W1cm,
    const float* __restrict__ b1, u16* __restrict__ h) {
  int lane = threadIdx.x & 31, wave = threadIdx.x >> 5;
  int tile = blockIdx.x * 4 + wave;
  int row0 = tile * 16;
  int m = lane & 15, hi = lane >> 4;
  FragBF a = load_a_bf16(rbf, (size_t)(row0 + m) * KRBF, 0, hi);
#pragma unroll
  for (int nt = 0; nt < 4; ++nt) {
    FragBF b = load_b_cm(W1cm, nt * 16 + m, KRBF, 0, hi);
    v8f c = {};
    c = wmma_bf16(a, b, c);
    float bias = b1[nt * 16 + m];
#pragma unroll
    for (int r = 0; r < 8; ++r) {
      int mm = r + (hi << 3);
      h[(size_t)(row0 + mm) * 64 + nt * 16 + m] = bf16_rne(ssp(c[r] + bias));
    }
  }
}

// w = h @ W2 + b2, fused with message formation + atomic scatter.
// Block = 128 threads = 4 waves = 64 edges. Wave handles a 16-edge tile.
__global__ __launch_bounds__(128) void k_edge_msg(
    const u16* __restrict__ h, const u16* __restrict__ W2cm,
    const float* __restrict__ b2, const int* __restrict__ eidx,
    const float* __restrict__ y1,
    const float* __restrict__ s, const float* __restrict__ vv,
    float* __restrict__ a_s, float* __restrict__ a_v) {
  __shared__ int   sh_src[64], sh_dst[64];
  __shared__ float sh_y[64][3];
  int t = threadIdx.x;
  int eb0 = blockIdx.x * 64;
  if (t < 64) {
    int e = eb0 + t;
    sh_src[t] = eidx[e];
    sh_dst[t] = eidx[NEDGES + e];
    sh_y[t][0] = y1[e * 3 + 0];
    sh_y[t][1] = y1[e * 3 + 1];
    sh_y[t][2] = y1[e * 3 + 2];
  }
  __syncthreads();
  int lane = t & 31, wave = t >> 5;
  int m = lane & 15, hi = lane >> 4;
  int tbase = wave * 16;         // LDS-local edge tile base
  int row0 = eb0 + tbase;        // global edge row base
  FragBF af0 = load_a_bf16(h, (size_t)(row0 + m) * 64, 0, hi);
  FragBF af1 = load_a_bf16(h, (size_t)(row0 + m) * 64, 32, hi);
#pragma unroll 1
  for (int cb = 0; cb < 8; ++cb) {
    v8f acc[5];
#pragma unroll
    for (int comp = 0; comp < 5; ++comp) {
      int col = comp * CCH + cb * 16 + m;       // n_local = lane&15
      float bias = b2[col];
      v8f c = {bias, bias, bias, bias, bias, bias, bias, bias};
      FragBF b0 = load_b_cm(W2cm, col, 64, 0, hi);
      c = wmma_bf16(af0, b0, c);
      FragBF b1f = load_b_cm(W2cm, col, 64, 32, hi);
      c = wmma_bf16(af1, b1f, c);
      acc[comp] = c;
    }
    int ch = cb * 16 + m;                        // global channel
#pragma unroll
    for (int r = 0; r < 8; ++r) {
      int li = tbase + r + (hi << 3);
      int si = sh_src[li], di = sh_dst[li];
      float yx = sh_y[li][0], yy = sh_y[li][1], yz = sh_y[li][2];
      float ss0 = s[(size_t)si * CCH + ch];
      const float* vp = vv + (size_t)si * (3 * CCH) + ch * 3;
      float v0 = vp[0], v1 = vp[1], v2 = vp[2];
      float wA = acc[0][r], wB = acc[1][r], wC = acc[2][r];
      float wD = acc[3][r], wE = acc[4][r];
      float dvy = v0 * yx + v1 * yy + v2 * yz;
      float ms  = wA * ss0 + wB * dvy;
      float cx = v1 * yz - v2 * yy;
      float cy = v2 * yx - v0 * yz;
      float cz = v0 * yy - v1 * yx;
      float mvx = wC * ss0 * yx + wD * v0 + wE * cx;
      float mvy = wC * ss0 * yy + wD * v1 + wE * cy;
      float mvz = wC * ss0 * yz + wD * v2 + wE * cz;
      atomicAdd(&a_s[(size_t)di * CCH + ch], ms);
      float* ap = a_v + (size_t)di * (3 * CCH) + ch * 3;
      atomicAdd(ap + 0, mvx);
      atomicAdd(ap + 1, mvy);
      atomicAdd(ap + 2, mvz);
    }
  }
}

// Node update: s2 = a_s@Ws+bs ; gate = ssp(a_s@Wg+bg) ; v2 = a_v@Wv (per comp)
// Ws + Wg (64 KB) staged into LDS by the Tensor Data Mover once per block;
// WMMA B-fragments for them come from LDS (ds_load_b128), Wv from global.
// s,v updated in place (residual from layer > 0). One wave per 16-node tile.
__global__ __launch_bounds__(128) void k_node_update(
    const float* __restrict__ a_s, const float* __restrict__ a_v,
    const float* __restrict__ inv_cnt,
    const u16* __restrict__ Wscm, const float* __restrict__ bs,
    const u16* __restrict__ Wvcm,
    const u16* __restrict__ Wgcm, const float* __restrict__ bg,
    float* __restrict__ s, float* __restrict__ vv, int addRes) {
  __shared__ u16 sW[2 * CCH * CCH];   // [0..16383]=Ws, [16384..]=Wg (64 KB)
  int t = threadIdx.x;
#if USE_TDM
  if (t < 32) {                        // wave 0 drives the TDM
    u32 base = (u32)(uintptr_t)(&sW[0]);
    tdm_load_1d(Wscm, base, CCH * CCH);
    tdm_load_1d(Wgcm, base + CCH * CCH * 2, CCH * CCH);
    __builtin_amdgcn_s_wait_tensorcnt(0);
  }
#else
  for (int i = t; i < (CCH * CCH) / 8; i += 128) {
    ((uint4*)sW)[i] = ((const uint4*)Wscm)[i];
    ((uint4*)(sW + CCH * CCH))[i] = ((const uint4*)Wgcm)[i];
  }
#endif
  __syncthreads();

  int lane = t & 31, wave = t >> 5;
  int tile = blockIdx.x * 4 + wave;
  int row0 = tile * 16;
  int m = lane & 15, hi = lane >> 4;
  int node_m = row0 + m;
  float ic = inv_cnt[node_m];
  FragBF As[4], Av0[4], Av1[4], Av2[4];
  const float* srow = a_s + (size_t)node_m * CCH;
  const float* vrow = a_v + (size_t)node_m * (3 * CCH);
#pragma unroll
  for (int f = 0; f < 4; ++f) {
    As[f]  = load_a_f32(srow,     f * 32, hi, ic, 1);
    Av0[f] = load_a_f32(vrow + 0, f * 32, hi, ic, 3);
    Av1[f] = load_a_f32(vrow + 1, f * 32, hi, ic, 3);
    Av2[f] = load_a_f32(vrow + 2, f * 32, hi, ic, 3);
  }
#pragma unroll 1
  for (int ct = 0; ct < 8; ++ct) {
    int col = ct * 16 + m;
    float b_s = bs[col], b_g = bg[col];
    v8f cS = {b_s, b_s, b_s, b_s, b_s, b_s, b_s, b_s};
    v8f cG = {b_g, b_g, b_g, b_g, b_g, b_g, b_g, b_g};
    v8f cV0 = {}, cV1 = {}, cV2 = {};
#pragma unroll
    for (int f = 0; f < 4; ++f) {
      FragBF bS = load_b_cm(sW, col, CCH, f * 32, hi);               // LDS
      cS = wmma_bf16(As[f], bS, cS);
      FragBF bG = load_b_cm(sW + CCH * CCH, col, CCH, f * 32, hi);   // LDS
      cG = wmma_bf16(As[f], bG, cG);
      FragBF bV = load_b_cm(Wvcm, col, CCH, f * 32, hi);             // global
      cV0 = wmma_bf16(Av0[f], bV, cV0);
      cV1 = wmma_bf16(Av1[f], bV, cV1);
      cV2 = wmma_bf16(Av2[f], bV, cV2);
    }
#pragma unroll
    for (int r = 0; r < 8; ++r) {
      int node = row0 + r + (hi << 3);
      float gate = ssp(cG[r]);
      float sn = ssp(cS[r]);
      float n0 = cV0[r] * gate, n1 = cV1[r] * gate, n2 = cV2[r] * gate;
      size_t siA = (size_t)node * CCH + col;
      size_t viA = (size_t)node * (3 * CCH) + col * 3;
      if (addRes) {
        sn += s[siA];
        n0 += vv[viA + 0]; n1 += vv[viA + 1]; n2 += vv[viA + 2];
      }
      s[siA] = sn;
      vv[viA + 0] = n0; vv[viA + 1] = n1; vv[viA + 2] = n2;
    }
  }
}

__global__ void k_finalize(const float* __restrict__ s, const float* __restrict__ vv,
                           float* __restrict__ out) {
  int id = blockIdx.x * blockDim.x + threadIdx.x;
  int n = id >> 7, c = id & 127;
  if (n >= NNODES) return;
  out[(size_t)n * 512 + c] = s[id];
  const float* vp = vv + (size_t)n * (3 * CCH) + c * 3;
  float* op = out + (size_t)n * 512 + CCH + c * 3;
  op[0] = vp[0]; op[1] = vp[1]; op[2] = vp[2];
}

// ------------------------- host launcher -----------------------------------

extern "C" void kernel_launch(void* const* d_in, const int* in_sizes, int n_in,
                              void* d_out, int out_size, void* d_ws, size_t ws_size,
                              hipStream_t stream) {
  const float *pos, *embed, *alpha_raw;
  const int *an, *ei;
  struct Lay { const float *W1, *b1, *W2, *b2, *Ws, *bs, *Wv, *Wg, *bg; } L[NLAYER];

  if (in_sizes[0] == 3 * NNODES) {
    // insertion order: pos, atomic_numbers, edge_index, embed, alpha_raw,
    //                  then 5 x (W1,b1,W2,b2,Ws,bs,Wv,Wg,bg)
    pos = (const float*)d_in[0]; an = (const int*)d_in[1];
    ei = (const int*)d_in[2]; embed = (const float*)d_in[3];
    alpha_raw = (const float*)d_in[4];
    for (int l = 0; l < NLAYER; ++l) {
      int b = 5 + l * 9;
      L[l] = { (const float*)d_in[b+0], (const float*)d_in[b+1],
               (const float*)d_in[b+2], (const float*)d_in[b+3],
               (const float*)d_in[b+4], (const float*)d_in[b+5],
               (const float*)d_in[b+6], (const float*)d_in[b+7],
               (const float*)d_in[b+8] };
    }
  } else {
    // jax sorted-leaf order: atomic_numbers, edge_index, alpha_raw, embed,
    //                        5 x (W1,W2,Wg,Ws,Wv,b1,b2,bg,bs), pos
    an = (const int*)d_in[0]; ei = (const int*)d_in[1];
    alpha_raw = (const float*)d_in[2]; embed = (const float*)d_in[3];
    for (int l = 0; l < NLAYER; ++l) {
      int b = 4 + l * 9;
      L[l].W1 = (const float*)d_in[b+0]; L[l].W2 = (const float*)d_in[b+1];
      L[l].Wg = (const float*)d_in[b+2]; L[l].Ws = (const float*)d_in[b+3];
      L[l].Wv = (const float*)d_in[b+4]; L[l].b1 = (const float*)d_in[b+5];
      L[l].b2 = (const float*)d_in[b+6]; L[l].bg = (const float*)d_in[b+7];
      L[l].bs = (const float*)d_in[b+8];
    }
    pos = (const float*)d_in[4 + NLAYER * 9];
  }

  // workspace carve-up
  uint8_t* w = (uint8_t*)d_ws;
  size_t off = 0;
  auto alloc = [&](size_t bytes) -> void* {
    void* p = w + off;
    off = (off + bytes + 255) & ~(size_t)255;
    return p;
  };
  u16*   rbf  = (u16*)  alloc((size_t)NEDGES * KRBF * 2);
  float* y1   = (float*)alloc((size_t)NEDGES * 3 * 4);
  u16*   h    = (u16*)  alloc((size_t)NEDGES * 64 * 2);
  float* sB   = (float*)alloc((size_t)NNODES * CCH * 4);
  float* vB   = (float*)alloc((size_t)NNODES * 3 * CCH * 4);
  float* aS   = (float*)alloc((size_t)NNODES * CCH * 4);
  float* aV   = (float*)alloc((size_t)NNODES * 3 * CCH * 4);
  float* cnt  = (float*)alloc((size_t)NNODES * 4);
  float* invc = (float*)alloc((size_t)NNODES * 4);
  u16 *W1cm[NLAYER], *W2cm[NLAYER], *Wscm[NLAYER], *Wvcm[NLAYER], *Wgcm[NLAYER];
  for (int l = 0; l < NLAYER; ++l) {
    W1cm[l] = (u16*)alloc((size_t)KRBF * 64 * 2);
    W2cm[l] = (u16*)alloc((size_t)64 * 640 * 2);
    Wscm[l] = (u16*)alloc((size_t)CCH * CCH * 2);
    Wvcm[l] = (u16*)alloc((size_t)CCH * CCH * 2);
    Wgcm[l] = (u16*)alloc((size_t)CCH * CCH * 2);
  }

  (void)hipMemsetAsync(cnt, 0, (size_t)NNODES * 4, stream);
  (void)hipMemsetAsync(vB, 0, (size_t)NNODES * 3 * CCH * 4, stream);

  for (int l = 0; l < NLAYER; ++l) {
    k_repack_cm<<<(KRBF * 64 + 255) / 256, 256, 0, stream>>>(L[l].W1, W1cm[l], KRBF, 64);
    k_repack_cm<<<(64 * 640 + 255) / 256, 256, 0, stream>>>(L[l].W2, W2cm[l], 64, 640);
    k_repack_cm<<<(CCH * CCH + 255) / 256, 256, 0, stream>>>(L[l].Ws, Wscm[l], CCH, CCH);
    k_repack_cm<<<(CCH * CCH + 255) / 256, 256, 0, stream>>>(L[l].Wv, Wvcm[l], CCH, CCH);
    k_repack_cm<<<(CCH * CCH + 255) / 256, 256, 0, stream>>>(L[l].Wg, Wgcm[l], CCH, CCH);
  }

  k_edge_geom<<<NEDGES / 256, 256, 0, stream>>>(pos, ei, alpha_raw, rbf, y1);
  k_count<<<NEDGES / 256, 256, 0, stream>>>(ei, cnt);
  k_invcnt<<<NNODES / 256, 256, 0, stream>>>(cnt, invc);
  k_init_s<<<(NNODES * CCH) / 256, 256, 0, stream>>>(embed, an, sB);

  for (int l = 0; l < NLAYER; ++l) {
    (void)hipMemsetAsync(aS, 0, (size_t)NNODES * CCH * 4, stream);
    (void)hipMemsetAsync(aV, 0, (size_t)NNODES * 3 * CCH * 4, stream);
    k_edge_mlp1<<<NEDGES / 64, 128, 0, stream>>>(rbf, W1cm[l], L[l].b1, h);
    k_edge_msg<<<NEDGES / 64, 128, 0, stream>>>(h, W2cm[l], L[l].b2, ei, y1,
                                                sB, vB, aS, aV);
    k_node_update<<<NNODES / 64, 128, 0, stream>>>(aS, aV, invc,
                                                   Wscm[l], L[l].bs, Wvcm[l],
                                                   Wgcm[l], L[l].bg,
                                                   sB, vB, (l > 0) ? 1 : 0);
  }
  k_finalize<<<(NNODES * CCH) / 256, 256, 0, stream>>>(sB, vB, (float*)d_out);
}
